// PETFullModel_30571577213363
// MI455X (gfx1250) — matrix operand order repaired and checked by reading, not compiled, not run on hardware
//
#include <hip/hip_runtime.h>
#include <hip/hip_bf16.h>

typedef _Float16 half_t;
typedef __attribute__((ext_vector_type(16))) _Float16 v16h;
typedef __attribute__((ext_vector_type(8)))  _Float16 v8h;
typedef __attribute__((ext_vector_type(8)))  float    v8f;

#define NN  8192
#define MM  32
#define DD  256
#define LL  2
#define HH  8
#define SS  119
#define DFN 1024
#define DHN 32
#define TOK 33
#define TPD 48

// Dynamic LDS carve (bytes). gfx1250 WGP has 320KB; we use ~153KB/WG.
#define OFF_TOK 0            // float  33*256          = 33792 B
#define OFF_X   33792        // half   48*256          = 24576 B
#define OFF_QC  58368        // half   48*256 (q -> attn-out -> edge silu)
#define OFF_KT  82944        // half   256*48 (kT; reused as FFN h1 chunk)
#define OFF_V   107520       // half   64*256 (rows 48..63 zero)
#define OFF_AT  140288       // half   8 waves * 16*64 attn scratch (+red alias)
#define SMEM_BYTES 156672

// ---------------------------------------------------------------------------
// WMMA helpers (CDNA5 gfx1250, wave32). Layouts per CDNA5 ISA.
// ---------------------------------------------------------------------------
__device__ __forceinline__ v8f wmma_f16(v16h a, v16h b, v8f c) {
  return __builtin_amdgcn_wmma_f32_16x16x32_f16(false, a, false, b,
                                                (short)0, c, false, false);
}

// A-fragment: 16x32 f16. lane<16: M=lane, kb=0; lane>=16: M=lane-16, kb=8.
// Halves = {K=kb..kb+7, K=kb+16..kb+23}: two contiguous 16B groups.
__device__ __forceinline__ v16h ldsA(const half_t* buf, int row0, int k0,
                                     int ld, int lane) {
  int m  = lane & 15;
  int kb = (lane >> 4) << 3;
  const half_t* p = buf + (size_t)(row0 + m) * ld + k0 + kb;
  v8h lo = *(const v8h*)p;
  v8h hi = *(const v8h*)(p + 16);
  return __builtin_shufflevector(lo, hi, 0, 1, 2, 3, 4, 5, 6, 7,
                                 8, 9, 10, 11, 12, 13, 14, 15);
}

// B-fragment: 32x16 f16. N = lane&15; K = kb..kb+7, kb+16..kb+23 (kb = 8*(lane>>4)).
__device__ __forceinline__ v16h ldB(const half_t* buf, int k0, int n0,
                                    int ld, int lane) {
  int nn = n0 + (lane & 15);
  int kb = k0 + ((lane >> 4) << 3);
  v16h r;
#pragma unroll
  for (int i = 0; i < 8; ++i) {
    r[i]     = buf[(size_t)(kb + i) * ld + nn];
    r[8 + i] = buf[(size_t)(kb + 16 + i) * ld + nn];
  }
  return r;
}

// D store (f32->f16): element i -> (row0 + 8*(lane>>4) + i, col0 + (lane&15))
__device__ __forceinline__ void stD_h(half_t* buf, v8f acc, int row0, int col0,
                                      int ld, int lane) {
  int c  = col0 + (lane & 15);
  int rb = row0 + ((lane >> 4) << 3);
#pragma unroll
  for (int i = 0; i < 8; ++i) buf[(size_t)(rb + i) * ld + c] = (half_t)acc[i];
}

// ---------------------------------------------------------------------------
struct P {
  const int*   species;
  const int*   pmask;
  const float* ev;
  const float* ed;
  const float* cf;
  const float* emb_node;
  const float* Wr; const float* br;
  const float* b1; const float* b2;
  const float* ln1g; const float* ln1b; const float* ln2g; const float* ln2b;
  const float* nhW; const float* nhb; const float* nfW; const float* nfb;
  const float* ehb; const float* efW; const float* efb;
  const half_t* WqH; const half_t* WkH; const half_t* WvH; const half_t* WoH;
  const half_t* W1H; const half_t* W2H; const half_t* ehWH;
  float* msgs;
  float* tok;
  float* out;
};

__device__ __forceinline__ void ln_rows(const float* stok, const float* g,
                                        const float* b, half_t* dst,
                                        int wid, int lane) {
  for (int r = wid; r < TOK; r += 8) {
    const float* row = stok + (size_t)r * DD;
    float s = 0.f, s2 = 0.f;
    for (int c = lane; c < DD; c += 32) {
      float v = row[c];
      s += v; s2 += v * v;
    }
#pragma unroll
    for (int o = 16; o > 0; o >>= 1) {
      s  += __shfl_xor(s, o, 32);
      s2 += __shfl_xor(s2, o, 32);
    }
    float mu  = s * (1.f / DD);
    float var = s2 * (1.f / DD) - mu * mu;
    float rs  = rsqrtf(var + 1e-5f);
    for (int c = lane; c < DD; c += 32)
      dst[(size_t)r * DD + c] = (half_t)((row[c] - mu) * rs * g[c] + b[c]);
  }
}

// ---------------------------------------------------------------------------
// One workgroup == one node, one layer.  256 threads = 8 waves (wave32).
// Residual stream + attention state in LDS; each global weight B-fragment
// is loaded exactly once per wave and applied to all 3 row-tiles.
// ---------------------------------------------------------------------------
__global__ __launch_bounds__(256) void pet_layer(P p, int l) {
  const int n    = blockIdx.x;
  const int tid  = threadIdx.x;
  const int lane = tid & 31;
  const int wid  = tid >> 5;

  extern __shared__ char smem[];
  float*  s_tok = (float*)(smem + OFF_TOK);
  half_t* s_x   = (half_t*)(smem + OFF_X);
  half_t* s_qc  = (half_t*)(smem + OFF_QC);
  half_t* s_kT  = (half_t*)(smem + OFF_KT);   // also FFN h1 chunk
  half_t* s_v   = (half_t*)(smem + OFF_V);
  half_t* s_at  = (half_t*)(smem + OFF_AT);
  float*  s_red = (float*)(smem + OFF_AT);    // used only in final reduce

  // ---- Phase 0: build tokens in LDS -------------------------------------
  {
    int sp = p.species[n];
    const float* ne  = p.emb_node + ((size_t)l * SS + sp) * DD;
    const float* WrL = p.Wr + (size_t)l * 4 * DD;
    const float* brL = p.br + (size_t)l * DD;
    for (int idx = tid; idx < TOK * DD; idx += 256) {
      int r = idx >> 8, c = idx & 255;
      float v;
      if (r == 0) {
        v = ne[c];
      } else {
        size_t e = (size_t)n * MM + (r - 1);
        float g0 = p.ev[e * 3 + 0], g1 = p.ev[e * 3 + 1];
        float g2 = p.ev[e * 3 + 2], g3 = p.ed[e];
        float eg = g0 * WrL[c] + g1 * WrL[DD + c] + g2 * WrL[2 * DD + c] +
                   g3 * WrL[3 * DD + c] + brL[c];
        v = (p.msgs[e * DD + c] + eg) * p.cf[e];
      }
      s_tok[idx] = v;
    }
    for (int idx = tid; idx < (TPD - TOK) * DD; idx += 256)
      s_x[TOK * DD + idx] = (half_t)0.f;
    for (int idx = tid; idx < 16 * DD; idx += 256)
      s_v[48 * DD + idx] = (half_t)0.f;
    __syncthreads();
  }

  // ---- LN1 -> s_x -------------------------------------------------------
  ln_rows(s_tok, p.ln1g + (size_t)l * DD, p.ln1b + (size_t)l * DD, s_x, wid, lane);
  __syncthreads();

  // ---- QKV for ALL heads; B loaded once, applied to 3 row-tiles ---------
  const half_t* Wq = p.WqH + (size_t)l * DD * DD;
  const half_t* Wk = p.WkH + (size_t)l * DD * DD;
  const half_t* Wv = p.WvH + (size_t)l * DD * DD;
  if (lane == 0) __builtin_prefetch(Wq + wid * 32, 0, 1);   // global_prefetch_b8

  for (int mat = 0; mat < 3; ++mat) {
    const half_t* W = (mat == 0) ? Wq : (mat == 1) ? Wk : Wv;
#pragma unroll
    for (int cc = 0; cc < 2; ++cc) {
      int ct = wid * 2 + cc;
      v8f a0 = {}, a1 = {}, a2 = {};
#pragma unroll
      for (int kk = 0; kk < 8; ++kk) {
        v16h b = ldB(W, kk * 32, ct * 16, DD, lane);
        a0 = wmma_f16(ldsA(s_x, 0,  kk * 32, DD, lane), b, a0);
        a1 = wmma_f16(ldsA(s_x, 16, kk * 32, DD, lane), b, a1);
        a2 = wmma_f16(ldsA(s_x, 32, kk * 32, DD, lane), b, a2);
      }
      if (mat == 0) {
        stD_h(s_qc, a0, 0, ct * 16, DD, lane);
        stD_h(s_qc, a1, 16, ct * 16, DD, lane);
        stD_h(s_qc, a2, 32, ct * 16, DD, lane);
      } else if (mat == 2) {
        stD_h(s_v, a0, 0, ct * 16, DD, lane);
        stD_h(s_v, a1, 16, ct * 16, DD, lane);
        stD_h(s_v, a2, 32, ct * 16, DD, lane);
      } else {   // k stored transposed: kT[dh_global][key], stride TPD
        int c  = ct * 16 + (lane & 15);
        int rb = (lane >> 4) << 3;
#pragma unroll
        for (int i = 0; i < 8; ++i) {
          s_kT[c * TPD + rb + i]      = (half_t)a0[i];
          s_kT[c * TPD + 16 + rb + i] = (half_t)a1[i];
          s_kT[c * TPD + 32 + rb + i] = (half_t)a2[i];
        }
      }
    }
  }
  __syncthreads();

  // ---- Attention: 24 (head, row-tile) jobs over 8 waves -----------------
  const float inv_sqrt_dh = 0.1767766952966369f;
  half_t* s_atw = s_at + wid * (16 * 64);       // per-wave 16x64 attn scratch
  for (int j = wid; j < 24; j += 8) {
    int h = j / 3, rt = j % 3;
    v16h aq = ldsA(s_qc, rt * 16, h * DHN, DD, lane);
    const half_t* kTh = s_kT + (size_t)h * DHN * TPD;
    v8f sc0 = wmma_f16(aq, ldB(kTh, 0, 0, TPD, lane), (v8f){});
    v8f sc1 = wmma_f16(aq, ldB(kTh, 0, 16, TPD, lane), (v8f){});
    v8f sc2 = wmma_f16(aq, ldB(kTh, 0, 32, TPD, lane), (v8f){});

    int key_lo = lane & 15;
    bool ok0 = (key_lo == 0) || (p.pmask[(size_t)n * MM + key_lo - 1] != 0);
    bool ok1 = (p.pmask[(size_t)n * MM + 15 + key_lo] != 0);
    bool ok2 = (key_lo == 0) && (p.pmask[(size_t)n * MM + 31] != 0);
#pragma unroll
    for (int i = 0; i < 8; ++i) {
      sc0[i] = ok0 ? sc0[i] * inv_sqrt_dh : -1e9f;
      sc1[i] = ok1 ? sc1[i] * inv_sqrt_dh : -1e9f;
      sc2[i] = ok2 ? sc2[i] * inv_sqrt_dh : -1e9f;
    }
#pragma unroll
    for (int i = 0; i < 8; ++i) {                 // row softmax in registers
      float mx = fmaxf(fmaxf(sc0[i], sc1[i]), sc2[i]);
#pragma unroll
      for (int o = 1; o < 16; o <<= 1) mx = fmaxf(mx, __shfl_xor(mx, o, 32));
      float e0 = __expf(sc0[i] - mx);
      float e1 = __expf(sc1[i] - mx);
      float e2 = __expf(sc2[i] - mx);
      float sm = e0 + e1 + e2;
#pragma unroll
      for (int o = 1; o < 16; o <<= 1) sm += __shfl_xor(sm, o, 32);
      float inv = 1.f / sm;
      sc0[i] = e0 * inv; sc1[i] = e1 * inv; sc2[i] = e2 * inv;
    }
    int rb = (lane >> 4) << 3;
#pragma unroll
    for (int i = 0; i < 8; ++i) {
      s_atw[(rb + i) * 64 + 0  + key_lo] = (half_t)sc0[i];
      s_atw[(rb + i) * 64 + 16 + key_lo] = (half_t)sc1[i];
      s_atw[(rb + i) * 64 + 32 + key_lo] = (half_t)sc2[i];
      s_atw[(rb + i) * 64 + 48 + key_lo] = (half_t)0.f;
    }
    v8f oh0 = {}, oh1 = {};
#pragma unroll
    for (int kk = 0; kk < 2; ++kk) {
      v16h a = ldsA(s_atw, 0, kk * 32, 64, lane);
      oh0 = wmma_f16(a, ldB(s_v, kk * 32, h * DHN + 0, DD, lane), oh0);
      oh1 = wmma_f16(a, ldB(s_v, kk * 32, h * DHN + 16, DD, lane), oh1);
    }
    stD_h(s_qc, oh0, rt * 16, h * DHN + 0, DD, lane);   // overwrite own q blk
    stD_h(s_qc, oh1, rt * 16, h * DHN + 16, DD, lane);
  }
  __syncthreads();

  // ---- o @ Wo + residual into s_tok (B loaded once) ---------------------
  const half_t* Wo = p.WoH + (size_t)l * DD * DD;
#pragma unroll
  for (int cc = 0; cc < 2; ++cc) {
    int ct = wid * 2 + cc;
    v8f a0 = {}, a1 = {}, a2 = {};
#pragma unroll
    for (int kk = 0; kk < 8; ++kk) {
      v16h b = ldB(Wo, kk * 32, ct * 16, DD, lane);
      a0 = wmma_f16(ldsA(s_qc, 0,  kk * 32, DD, lane), b, a0);
      a1 = wmma_f16(ldsA(s_qc, 16, kk * 32, DD, lane), b, a1);
      a2 = wmma_f16(ldsA(s_qc, 32, kk * 32, DD, lane), b, a2);
    }
    int c  = ct * 16 + (lane & 15);
    int rb = (lane >> 4) << 3;
#pragma unroll
    for (int i = 0; i < 8; ++i) {
      int r0 = rb + i;
      s_tok[(size_t)r0 * DD + c] += a0[i];
      s_tok[(size_t)(16 + r0) * DD + c] += a1[i];
      if (32 + r0 < TOK) s_tok[(size_t)(32 + r0) * DD + c] += a2[i];
    }
  }
  __syncthreads();

  // ---- LN2 -> s_x (pad rows still zero) ---------------------------------
  ln_rows(s_tok, p.ln2g + (size_t)l * DD, p.ln2b + (size_t)l * DD, s_x, wid, lane);
  __syncthreads();

  // ---- FFN: split-K over 4 chunks of 256 through s_kT -------------------
  const half_t* W1 = p.W1H + (size_t)l * DD * DFN;
  const half_t* W2 = p.W2H + (size_t)l * DFN * DD;
  v8f o2[6];
#pragma unroll
  for (int i = 0; i < 6; ++i) o2[i] = (v8f){};
  for (int ch = 0; ch < 4; ++ch) {
#pragma unroll
    for (int cc = 0; cc < 2; ++cc) {
      int ct = wid * 2 + cc;
      v8f a0 = {}, a1 = {}, a2 = {};
#pragma unroll
      for (int kk = 0; kk < 8; ++kk) {
        v16h b = ldB(W1, kk * 32, ch * 256 + ct * 16, DFN, lane);
        a0 = wmma_f16(ldsA(s_x, 0,  kk * 32, DD, lane), b, a0);
        a1 = wmma_f16(ldsA(s_x, 16, kk * 32, DD, lane), b, a1);
        a2 = wmma_f16(ldsA(s_x, 32, kk * 32, DD, lane), b, a2);
      }
      float bias = p.b1[(size_t)l * DFN + ch * 256 + ct * 16 + (lane & 15)];
#pragma unroll
      for (int i = 0; i < 8; ++i) {
        float x0 = a0[i] + bias; a0[i] = x0 / (1.f + __expf(-x0));
        float x1 = a1[i] + bias; a1[i] = x1 / (1.f + __expf(-x1));
        float x2 = a2[i] + bias; a2[i] = x2 / (1.f + __expf(-x2));
      }
      stD_h(s_kT, a0, 0, ct * 16, DD, lane);
      stD_h(s_kT, a1, 16, ct * 16, DD, lane);
      stD_h(s_kT, a2, 32, ct * 16, DD, lane);
    }
    __syncthreads();
#pragma unroll
    for (int cc = 0; cc < 2; ++cc) {
      int ct = wid * 2 + cc;
#pragma unroll
      for (int kk = 0; kk < 8; ++kk) {
        v16h b = ldB(W2, ch * 256 + kk * 32, ct * 16, DD, lane);
        o2[0 + cc] = wmma_f16(ldsA(s_kT, 0,  kk * 32, DD, lane), b, o2[0 + cc]);
        o2[2 + cc] = wmma_f16(ldsA(s_kT, 16, kk * 32, DD, lane), b, o2[2 + cc]);
        o2[4 + cc] = wmma_f16(ldsA(s_kT, 32, kk * 32, DD, lane), b, o2[4 + cc]);
      }
    }
    __syncthreads();
  }
#pragma unroll
  for (int rt = 0; rt < 3; ++rt)
#pragma unroll
    for (int cc = 0; cc < 2; ++cc) {
      int ct = wid * 2 + cc, ix = rt * 2 + cc;
      int c  = ct * 16 + (lane & 15);
      int rb = rt * 16 + ((lane >> 4) << 3);
      float bias = p.b2[(size_t)l * DD + c];
#pragma unroll
      for (int i = 0; i < 8; ++i) {
        int r = rb + i;
        if (r < TOK) s_tok[(size_t)r * DD + c] += o2[ix][i] + bias;
      }
    }
  __syncthreads();

  // ---- write residual stream back to global (for rev gather) ------------
  {
    float* tokg = p.tok + (size_t)n * TOK * DD;
    for (int idx = tid; idx < TOK * DD; idx += 256) tokg[idx] = s_tok[idx];
  }

  // ---- energy heads ------------------------------------------------------
  for (int idx = tid; idx < TPD * DD; idx += 256) {
    int r = idx >> 8, c = idx & 255;
    s_x[idx] = (r < 32) ? (half_t)s_tok[(size_t)(r + 1) * DD + c] : (half_t)0.f;
  }
  __syncthreads();
  const half_t* eW = p.ehWH + (size_t)l * DD * DD;
#pragma unroll
  for (int cc = 0; cc < 2; ++cc) {
    int ct = wid * 2 + cc;
    v8f a0 = {}, a1 = {};
#pragma unroll
    for (int kk = 0; kk < 8; ++kk) {
      v16h b = ldB(eW, kk * 32, ct * 16, DD, lane);
      a0 = wmma_f16(ldsA(s_x, 0,  kk * 32, DD, lane), b, a0);
      a1 = wmma_f16(ldsA(s_x, 16, kk * 32, DD, lane), b, a1);
    }
    float bias = p.ehb[(size_t)l * DD + ct * 16 + (lane & 15)];
#pragma unroll
    for (int i = 0; i < 8; ++i) {
      float x0 = a0[i] + bias; a0[i] = x0 / (1.f + __expf(-x0));
      float x1 = a1[i] + bias; a1[i] = x1 / (1.f + __expf(-x1));
    }
    stD_h(s_qc, a0, 0, ct * 16, DD, lane);
    stD_h(s_qc, a1, 16, ct * 16, DD, lane);
  }
  __syncthreads();

  float part = 0.f;
  for (int m = wid; m < 32; m += 8) {             // edge final matvec
    float s = 0.f;
    for (int c = lane; c < DD; c += 32)
      s += (float)s_qc[(size_t)m * DD + c] * p.efW[(size_t)l * DD + c];
#pragma unroll
    for (int o = 16; o > 0; o >>= 1) s += __shfl_xor(s, o, 32);
    if (lane == 0 && p.pmask[(size_t)n * MM + m] != 0)
      part += (s + p.efb[l]) * p.cf[(size_t)n * MM + m];
  }
  {                                               // node head: 1 col/thread
    int jv = tid;
    float a = p.nhb[(size_t)l * DD + jv];
    for (int c = 0; c < DD; ++c)
      a += s_tok[c] * p.nhW[((size_t)l * DD + c) * DD + jv];
    float sl = (a / (1.f + __expf(-a))) * p.nfW[(size_t)l * DD + jv];
#pragma unroll
    for (int o = 16; o > 0; o >>= 1) sl += __shfl_xor(sl, o, 32);
    if (lane == 0) part += sl;
  }
  __syncthreads();
  if (lane == 0) s_red[wid] = part;
  __syncthreads();
  if (tid == 0) {
    float e = p.nfb[l];
    for (int w = 0; w < 8; ++w) e += s_red[w];
    atomicAdd(p.out + n, e);
  }
}

// ---------------------------------------------------------------------------
__global__ void init_msgs_k(const int* nbr_sp, const float* emb_nb,
                            float* msgs) {
  size_t idx = (size_t)blockIdx.x * 256 + threadIdx.x;  // N*M*D
  size_t e = idx >> 8;
  int c = (int)(idx & 255);
  msgs[idx] = emb_nb[(size_t)nbr_sp[e] * DD + c];
}

__global__ void rev_msgs_k(const int* rev, const float* tok, float* msgs) {
  size_t idx = (size_t)blockIdx.x * 256 + threadIdx.x;  // N*M*D
  size_t e = idx >> 8;
  int c = (int)(idx & 255);
  int j  = rev[e];
  int nj = j >> 5;
  int mj = j & 31;
  float rv = tok[((size_t)nj * TOK + 1 + mj) * DD + c];
  msgs[idx] = 0.5f * (msgs[idx] + rv);
}

__global__ void f2h_k(const float* s, half_t* d, int nelem) {
  int i = blockIdx.x * 256 + threadIdx.x;
  if (i < nelem) d[i] = (half_t)s[i];
}

__global__ void zero_k(float* o, int nelem) {
  int i = blockIdx.x * 256 + threadIdx.x;
  if (i < nelem) o[i] = 0.f;
}

// ---------------------------------------------------------------------------
extern "C" void kernel_launch(void* const* d_in, const int* in_sizes, int n_in,
                              void* d_out, int out_size, void* d_ws,
                              size_t ws_size, hipStream_t stream) {
  const int*   species = (const int*)d_in[0];
  const int*   nbr_sp  = (const int*)d_in[1];
  const float* ev      = (const float*)d_in[2];
  const float* ed      = (const float*)d_in[3];
  const int*   pmask   = (const int*)d_in[4];
  const int*   rev     = (const int*)d_in[5];
  const float* cf      = (const float*)d_in[6];
  const float* emb_nb  = (const float*)d_in[7];
  const float* emb_nd  = (const float*)d_in[8];
  const float* Wr      = (const float*)d_in[9];
  const float* br      = (const float*)d_in[10];
  const float* Wq      = (const float*)d_in[11];
  const float* Wk      = (const float*)d_in[12];
  const float* Wv      = (const float*)d_in[13];
  const float* Wo      = (const float*)d_in[14];
  const float* W1      = (const float*)d_in[15];
  const float* b1      = (const float*)d_in[16];
  const float* W2      = (const float*)d_in[17];
  const float* b2      = (const float*)d_in[18];
  const float* ln1g    = (const float*)d_in[19];
  const float* ln1b    = (const float*)d_in[20];
  const float* ln2g    = (const float*)d_in[21];
  const float* ln2b    = (const float*)d_in[22];
  const float* nhW     = (const float*)d_in[23];
  const float* nhb     = (const float*)d_in[24];
  const float* nfW     = (const float*)d_in[25];
  const float* nfb     = (const float*)d_in[26];
  const float* ehW     = (const float*)d_in[27];
  const float* ehb     = (const float*)d_in[28];
  const float* efW     = (const float*)d_in[29];
  const float* efb     = (const float*)d_in[30];
  float* out = (float*)d_out;

  char* w = (char*)d_ws;
  float* msgs = (float*)w; w += (size_t)NN * MM * DD * 4;
  float* tok  = (float*)w; w += (size_t)NN * TOK * DD * 4;
  half_t* WqH = (half_t*)w; w += (size_t)LL * DD * DD * 2;
  half_t* WkH = (half_t*)w; w += (size_t)LL * DD * DD * 2;
  half_t* WvH = (half_t*)w; w += (size_t)LL * DD * DD * 2;
  half_t* WoH = (half_t*)w; w += (size_t)LL * DD * DD * 2;
  half_t* W1H = (half_t*)w; w += (size_t)LL * DD * DFN * 2;
  half_t* W2H = (half_t*)w; w += (size_t)LL * DFN * DD * 2;
  half_t* eWH = (half_t*)w; w += (size_t)LL * DD * DD * 2;

  const int nDD = LL * DD * DD, nDF = LL * DD * DFN;
  f2h_k<<<(nDD + 255) / 256, 256, 0, stream>>>(Wq, WqH, nDD);
  f2h_k<<<(nDD + 255) / 256, 256, 0, stream>>>(Wk, WkH, nDD);
  f2h_k<<<(nDD + 255) / 256, 256, 0, stream>>>(Wv, WvH, nDD);
  f2h_k<<<(nDD + 255) / 256, 256, 0, stream>>>(Wo, WoH, nDD);
  f2h_k<<<(nDF + 255) / 256, 256, 0, stream>>>(W1, W1H, nDF);
  f2h_k<<<(nDF + 255) / 256, 256, 0, stream>>>(W2, W2H, nDF);
  f2h_k<<<(nDD + 255) / 256, 256, 0, stream>>>(ehW, eWH, nDD);
  zero_k<<<(NN + 255) / 256, 256, 0, stream>>>(out, NN);

  const size_t emd = (size_t)NN * MM * DD;
  init_msgs_k<<<(unsigned)(emd / 256), 256, 0, stream>>>(nbr_sp, emb_nb, msgs);

  P p;
  p.species = species; p.pmask = pmask; p.ev = ev; p.ed = ed; p.cf = cf;
  p.emb_node = emb_nd; p.Wr = Wr; p.br = br; p.b1 = b1; p.b2 = b2;
  p.ln1g = ln1g; p.ln1b = ln1b; p.ln2g = ln2g; p.ln2b = ln2b;
  p.nhW = nhW; p.nhb = nhb; p.nfW = nfW; p.nfb = nfb;
  p.ehb = ehb; p.efW = efW; p.efb = efb;
  p.WqH = WqH; p.WkH = WkH; p.WvH = WvH; p.WoH = WoH;
  p.W1H = W1H; p.W2H = W2H; p.ehWH = eWH;
  p.msgs = msgs; p.tok = tok; p.out = out;

  static bool attr_done = []() {
    (void)hipFuncSetAttribute((const void*)pet_layer,
                              hipFuncAttributeMaxDynamicSharedMemorySize,
                              SMEM_BYTES);
    return true;
  }();
  (void)attr_done;

  for (int l = 0; l < LL; ++l) {
    pet_layer<<<NN, 256, SMEM_BYTES, stream>>>(p, l);
    if (l == 0)
      rev_msgs_k<<<(unsigned)(emd / 256), 256, 0, stream>>>(rev, tok, msgs);
  }
}
